// AttentionHead_26104811225428
// MI455X (gfx1250) — matrix operand order repaired and checked by reading, not compiled, not run on hardware
//
#include <hip/hip_runtime.h>
#include <hip/hip_bf16.h>
#include <stdint.h>

// Problem constants (match reference)
#define B_    4
#define S_    2048
#define DIN_  1024
#define D_    64      // DQ == DK == 64

typedef __attribute__((ext_vector_type(16))) _Float16 v16h;
typedef __attribute__((ext_vector_type(8)))  _Float16 v8h;
typedef __attribute__((ext_vector_type(8)))  float    v8f;

// Global-token mask exceptions: (row, col) pairs where the causal mask is
// forced to 0 even if col > row. These come from np.random.default_rng(0)
// .choice(2048, 3, replace=False) in the reference (fixed constants of the
// problem, not runtime inputs).
constexpr int GROW[3] = { 763, 1536, 901 };
constexpr int GCOL[3] = { 1157, 89, 1990 };

static __device__ __forceinline__ v8f wmma16(v16h a, v16h b, v8f c) {
  // D = A(16x32 f16) * B(32x16 f16) + C(16x16 f32)
  return __builtin_amdgcn_wmma_f32_16x16x32_f16(
      /*neg_a=*/false, a, /*neg_b=*/false, b,
      /*c_mod=*/(short)0, c, /*reuse_a=*/false, /*reuse_b=*/false);
}

static __device__ __forceinline__ v16h cat8(v8h lo, v8h hi) {
  return __builtin_shufflevector(lo, hi, 0, 1, 2, 3, 4, 5, 6, 7,
                                 8, 9, 10, 11, 12, 13, 14, 15);
}

static __device__ __forceinline__ v8f zero8() {
  v8f r;
#pragma unroll
  for (int i = 0; i < 8; ++i) r[i] = 0.0f;
  return r;
}

static __device__ __forceinline__ float redMax16(float v) {
  v = fmaxf(v, __shfl_xor(v, 1, 16));
  v = fmaxf(v, __shfl_xor(v, 2, 16));
  v = fmaxf(v, __shfl_xor(v, 4, 16));
  v = fmaxf(v, __shfl_xor(v, 8, 16));
  return v;
}

static __device__ __forceinline__ float redSum16(float v) {
  v += __shfl_xor(v, 1, 16);
  v += __shfl_xor(v, 2, 16);
  v += __shfl_xor(v, 4, 16);
  v += __shfl_xor(v, 8, 16);
  return v;
}

static __device__ __forceinline__ bool isGlobalTok(int row, int col) {
  bool g = false;
#pragma unroll
  for (int i = 0; i < 3; ++i) g |= (row == GROW[i] && col == GCOL[i]);
  return g;
}

// ---------------------------------------------------------------------------
// Projection: Out = f16( (X[M,1024] @ W[1024,64] + bias) * scale )
// One wave computes a 16x64 tile; 8 waves/block -> 128 rows per block.
// W slab staged TRANSPOSED (Wt[n][k], f16) in LDS so B-fragments are
// contiguous v16h LDS loads (ds_load_b128) instead of scalar gathers.
// vmode=1 writes the output tile-transposed: [row/32][feat(64)][row%32],
// which makes attention's P*V B-fragments contiguous global b128 loads.
// ---------------------------------------------------------------------------
__global__ __launch_bounds__(256) void proj_kernel(
    const float* __restrict__ X, const float* __restrict__ W,
    const float* __restrict__ bias, _Float16* __restrict__ Out,
    float scale, int vmode) {
  __shared__ _Float16 Wt[D_ * 128];  // transposed slab [n][k], 16 KB

  const int tid  = threadIdx.x;
  const int wv   = tid >> 5;
  const int lane = tid & 31;
  const int m    = lane & 15;   // row within 16-row tile (A / D layout)
  const int hi   = lane >> 4;   // lane half selects K/M sub-blocks

  const size_t row0 = ((size_t)blockIdx.x * 8 + wv) * 16;
  const float* xrow = X + (row0 + m) * DIN_;

  v8f c0 = zero8(), c1 = zero8(), c2 = zero8(), c3 = zero8();

  for (int kb = 0; kb < DIN_; kb += 128) {
    __syncthreads();
    // Transposed staging: coalesced f32 row reads, packed-f16 b32 LDS stores.
#pragma unroll
    for (int i = 0; i < 16; ++i) {
      const int p = tid + i * 256;  // pair index 0..4095
      const int n = p & 63;
      const int k = (p >> 6) * 2;
      union { uint32_t u; _Float16 h[2]; } pk;
      pk.h[0] = (_Float16)W[(size_t)(kb + k) * D_ + n];
      pk.h[1] = (_Float16)W[(size_t)(kb + k + 1) * D_ + n];
      *(uint32_t*)(&Wt[n * 128 + k]) = pk.u;
    }
    __syncthreads();

#pragma unroll
    for (int ksl = 0; ksl < 128; ksl += 32) {
      // A fragment: rows = m, K = kb+ksl+hi*16 .. +15 (contiguous K order);
      // one contiguous 16-float chunk -> b128 global loads + cvt_pk.
      v16h a;
#pragma unroll
      for (int e = 0; e < 16; ++e)
        a[e] = (_Float16)xrow[kb + ksl + hi * 16 + e];
      // B fragments: contiguous 16 halfs in Wt -> 2x ds_load_b128 each.
      const int ko = ksl + hi * 16;
      const v16h b0 = *(const v16h*)(&Wt[(0 * 16 + m) * 128 + ko]);
      const v16h b1 = *(const v16h*)(&Wt[(1 * 16 + m) * 128 + ko]);
      const v16h b2 = *(const v16h*)(&Wt[(2 * 16 + m) * 128 + ko]);
      const v16h b3 = *(const v16h*)(&Wt[(3 * 16 + m) * 128 + ko]);
      c0 = wmma16(a, b0, c0);
      c1 = wmma16(a, b1, c1);
      c2 = wmma16(a, b2, c2);
      c3 = wmma16(a, b3, c3);
    }
  }

  // Epilogue: D layout -> row = row0 + r + 8*hi, col = c*16 + m
#pragma unroll
  for (int r = 0; r < 8; ++r) {
    const size_t row = row0 + r + hi * 8;
    const _Float16 y0 = (_Float16)((c0[r] + bias[0  + m]) * scale);
    const _Float16 y1 = (_Float16)((c1[r] + bias[16 + m]) * scale);
    const _Float16 y2 = (_Float16)((c2[r] + bias[32 + m]) * scale);
    const _Float16 y3 = (_Float16)((c3[r] + bias[48 + m]) * scale);
    if (vmode) {
      // tile-transposed: [tile=row/32][feat][key=row%32]
      _Float16* t = Out + (row >> 5) * (size_t)(D_ * 32) + (row & 31);
      t[(0 * 16 + m) * 32] = y0;
      t[(1 * 16 + m) * 32] = y1;
      t[(2 * 16 + m) * 32] = y2;
      t[(3 * 16 + m) * 32] = y3;
    } else {
      Out[row * D_ + 0  + m] = y0;
      Out[row * D_ + 16 + m] = y1;
      Out[row * D_ + 32 + m] = y2;
      Out[row * D_ + 48 + m] = y3;
    }
  }
}

// ---------------------------------------------------------------------------
// Flash attention: per wave, one 16-row query tile, stream 32-key tiles with
// online softmax. Q is pre-scaled by 1/sqrt(d). K is row-major f16, V is
// tile-transposed f16 (so both feed WMMA with aligned b128 vector loads;
// both live in L2: 1 MB each). Only P needs LDS (cross-lane C->A transpose).
// ---------------------------------------------------------------------------
__global__ __launch_bounds__(256) void attn_kernel(
    const _Float16* __restrict__ Qh, const _Float16* __restrict__ Kh,
    const _Float16* __restrict__ Vt, float* __restrict__ Out) {
  __shared__ _Float16 Ps[8][16 * 32];  // 8 KB: per-wave P transpose buffer

  const int tid  = threadIdx.x;
  const int wv   = tid >> 5;
  const int lane = tid & 31;
  const int m    = lane & 15;
  const int hi   = lane >> 4;
  const int b    = blockIdx.y;
  const int qt   = blockIdx.x * 8 + wv;  // 16-row query tile, 0..127
  const int q0   = qt * 16;

  const _Float16* Qb = Qh + (size_t)b * S_ * D_;
  const _Float16* Kb = Kh + (size_t)b * S_ * D_;

  // Q A-fragments: 16 rows x 64 feats -> two K=32 fragments, vector loads.
  const _Float16* qrow = Qb + (size_t)(q0 + m) * D_ + hi * 16;
  const v16h aq0 = cat8(*(const v8h*)(qrow),      *(const v8h*)(qrow + 8));
  const v16h aq1 = cat8(*(const v8h*)(qrow + 32), *(const v8h*)(qrow + 40));

  // Online softmax state: lane holds rows r + 8*hi, r = 0..7
  float mrow[8], lrow[8];
#pragma unroll
  for (int r = 0; r < 8; ++r) { mrow[r] = -__builtin_inff(); lrow[r] = 0.0f; }
  v8f o0 = zero8(), o1 = zero8(), o2 = zero8(), o3 = zero8();

  auto process_tile = [&](int kt) {
    const int k0 = kt * 32;

    // Scores: S(16x32) = Q(16x64) * K^T; B[k][n] = K[k0+grp*16+n][k].
    // Contiguous 16 halfs per fragment -> global_load_b128 pairs.
    const _Float16* kr0 = Kb + (size_t)(k0 + m) * D_ + hi * 16;
    const _Float16* kr1 = kr0 + (size_t)16 * D_;
    const v16h b00 = *(const v16h*)(kr0);
    const v16h b01 = *(const v16h*)(kr0 + 32);
    const v16h b10 = *(const v16h*)(kr1);
    const v16h b11 = *(const v16h*)(kr1 + 32);
    v8f s0 = zero8(), s1 = zero8();
    s0 = wmma16(aq0, b00, s0);
    s0 = wmma16(aq1, b01, s0);
    s1 = wmma16(aq0, b10, s1);
    s1 = wmma16(aq1, b11, s1);

    // Mask + online softmax on f32 fragments.
    const int col0 = k0 + m;
    const int col1 = k0 + 16 + m;
#pragma unroll
    for (int r = 0; r < 8; ++r) {
      const int row = q0 + r + hi * 8;
      const bool keep0 = (col0 <= row) || isGlobalTok(row, col0);
      const bool keep1 = (col1 <= row) || isGlobalTok(row, col1);
      const float v0 = keep0 ? s0[r] : -__builtin_inff();
      const float v1 = keep1 ? s1[r] : -__builtin_inff();
      const float t = redMax16(fmaxf(v0, v1));
      const float mnew = fmaxf(mrow[r], t);
      const float alpha =
          (mnew == -__builtin_inff()) ? 1.0f : __expf(mrow[r] - mnew);
      const float p0 = keep0 ? __expf(v0 - mnew) : 0.0f;
      const float p1 = keep1 ? __expf(v1 - mnew) : 0.0f;
      mrow[r] = mnew;
      lrow[r] = lrow[r] * alpha + redSum16(p0 + p1);
      o0[r] *= alpha; o1[r] *= alpha; o2[r] *= alpha; o3[r] *= alpha;
      // Stash P in wave-private LDS to transpose C-layout -> A-layout.
      Ps[wv][(r + hi * 8) * 32 + m]      = (_Float16)p0;
      Ps[wv][(r + hi * 8) * 32 + 16 + m] = (_Float16)p1;
    }
    __builtin_amdgcn_wave_barrier();
    asm volatile("" ::: "memory");  // LDS ops are in-order per wave (DScnt)

    // P A-fragment (16x32): row m is contiguous in Ps -> 2x ds_load_b128.
    const _Float16* psrow = &Ps[wv][m * 32 + hi * 16];
    const v16h ap = cat8(*(const v8h*)(psrow), *(const v8h*)(psrow + 8));

    // V B-fragments from tile-transposed global V: B[k][n] = V[k0+k][c*16+n]
    // = Vt[tile][c*16+n][k], contiguous 16 halfs -> global_load_b128 pairs.
    const _Float16* vt =
        Vt + (((size_t)(b * S_ + k0)) >> 5) * (size_t)(D_ * 32) + hi * 16;
    const v16h bv0 = *(const v16h*)(vt + (0 * 16 + m) * 32);
    const v16h bv1 = *(const v16h*)(vt + (1 * 16 + m) * 32);
    const v16h bv2 = *(const v16h*)(vt + (2 * 16 + m) * 32);
    const v16h bv3 = *(const v16h*)(vt + (3 * 16 + m) * 32);
    o0 = wmma16(ap, bv0, o0);
    o1 = wmma16(ap, bv1, o1);
    o2 = wmma16(ap, bv2, o2);
    o3 = wmma16(ap, bv3, o3);
    __builtin_amdgcn_wave_barrier();
    asm volatile("" ::: "memory");
  };

  // Causal sweep: all 32-key tiles touching keys <= q0+15.
  const int ktend = (q0 + 15) >> 5;
  for (int kt = 0; kt <= ktend; ++kt) process_tile(kt);

  // Extra tiles for global tokens visible beyond the causal range.
  int extras[3]; int ne = 0;
#pragma unroll
  for (int g = 0; g < 3; ++g) {
    if (GROW[g] >= q0 && GROW[g] < q0 + 16) {
      const int kt = GCOL[g] >> 5;
      if (kt > ktend) {
        bool dup = false;
        for (int j = 0; j < ne; ++j) dup |= (extras[j] == kt);
        if (!dup) extras[ne++] = kt;
      }
    }
  }
  for (int j = 0; j < ne; ++j) process_tile(extras[j]);

  // Normalize and store f32 output [B,S,64].
  float* outb = Out + (size_t)b * S_ * D_;
#pragma unroll
  for (int r = 0; r < 8; ++r) {
    const float inv = 1.0f / lrow[r];
    const size_t row = (size_t)q0 + r + hi * 8;
    outb[row * D_ + 0  + m] = o0[r] * inv;
    outb[row * D_ + 16 + m] = o1[r] * inv;
    outb[row * D_ + 32 + m] = o2[r] * inv;
    outb[row * D_ + 48 + m] = o3[r] * inv;
  }
}

// ---------------------------------------------------------------------------
extern "C" void kernel_launch(void* const* d_in, const int* in_sizes, int n_in,
                              void* d_out, int out_size, void* d_ws,
                              size_t ws_size, hipStream_t stream) {
  (void)in_sizes; (void)n_in; (void)out_size; (void)ws_size;
  const float* q  = (const float*)d_in[0];
  const float* k  = (const float*)d_in[1];
  const float* v  = (const float*)d_in[2];
  const float* Wq = (const float*)d_in[3];
  const float* bq = (const float*)d_in[4];
  const float* Wk = (const float*)d_in[5];
  const float* bk = (const float*)d_in[6];
  const float* Wv = (const float*)d_in[7];
  const float* bv = (const float*)d_in[8];

  // Workspace: f16 Q (pre-scaled by 1/8), row-major K, tile-transposed V.
  _Float16* Qh  = (_Float16*)d_ws;
  _Float16* Kh  = Qh + (size_t)B_ * S_ * D_;
  _Float16* Vth = Kh + (size_t)B_ * S_ * D_;

  const int mrows = B_ * S_;          // 8192
  dim3 pgrid(mrows / 128);            // 64 blocks, 128 rows each
  proj_kernel<<<pgrid, 256, 0, stream>>>(q, Wq, bq, Qh, 0.125f, 0);
  proj_kernel<<<pgrid, 256, 0, stream>>>(k, Wk, bk, Kh, 1.0f, 0);
  proj_kernel<<<pgrid, 256, 0, stream>>>(v, Wv, bv, Vth, 1.0f, 1);

  dim3 agrid(S_ / 128, B_);           // 16 x 4 blocks, 8 query-tiles each
  attn_kernel<<<agrid, 256, 0, stream>>>(Qh, Kh, Vth, (float*)d_out);
}
